// MultiHeadAttention_78640851190221
// MI455X (gfx1250) — matrix operand order repaired and checked by reading, not compiled, not run on hardware
//
#include <hip/hip_runtime.h>

#define B_  4
#define N_  2048
#define D_  1024
#define H_  16
#define DK_ 64
#define BN_ (B_ * N_)

static constexpr float RSCALE = 0.125f;     // 1/sqrt(DK)
static constexpr float NEGBIG = -1.0e9f;

typedef unsigned short u16;
typedef __attribute__((ext_vector_type(16))) __bf16 v16bf;
typedef __attribute__((ext_vector_type(8)))  __bf16 v8bf;
typedef __attribute__((ext_vector_type(8)))  float  v8f;
typedef __attribute__((ext_vector_type(4)))  unsigned v4u;
typedef __attribute__((ext_vector_type(8)))  int    v8i;
typedef __attribute__((ext_vector_type(4)))  int    v4i;

#if __has_builtin(__builtin_amdgcn_tensor_load_to_lds) && \
    __has_builtin(__builtin_amdgcn_s_wait_tensorcnt)
#define USE_TDM 1
#else
#define USE_TDM 0
#endif

__device__ __forceinline__ u16 f2bf(float x) {
  union { float f; unsigned u; } c; c.f = x;
  unsigned u = c.u;
  u += 0x7FFFu + ((u >> 16) & 1u);            // round-to-nearest-even
  return (u16)(u >> 16);
}

// 16 contiguous bf16 -> B-fragment register group
__device__ __forceinline__ v16bf ldfrag16(const u16* p) {
  v8bf lo = *(const v8bf*)p;
  v8bf hi = *(const v8bf*)(p + 8);
  return __builtin_shufflevector(lo, hi, 0,1,2,3,4,5,6,7,8,9,10,11,12,13,14,15);
}
// A-fragment per-lane layout: elements [2*half8..2*half8+15] (contiguous)
__device__ __forceinline__ v16bf ldfragA(const u16* p, int half8) {
  v8bf lo = *(const v8bf*)(p + 2 * half8);
  v8bf hi = *(const v8bf*)(p + 2 * half8 + 8);
  return __builtin_shufflevector(lo, hi, 0,1,2,3,4,5,6,7,8,9,10,11,12,13,14,15);
}

#if USE_TDM
// Flat shared-pointer low 32 bits == LDS byte offset
__device__ __forceinline__ unsigned lds_off(const void* p) {
  return (unsigned)(size_t)p;
}
// Issue one 2D TDM tile load: tile_d1 rows x tile_d0 elems (bf16), row stride
// `stride` elems, packed tightly into LDS at lds byte offset `loff`.
// D# bitfield packing per CDNA5 ISA 8.3/8.4.
__device__ __forceinline__ void tdm_load_2d(unsigned loff, const void* gptr,
                                            unsigned tile_d0, unsigned tile_d1,
                                            unsigned tensor_d0, unsigned tensor_d1,
                                            unsigned stride) {
  unsigned long long ga = (unsigned long long)(size_t)gptr;
  v4u g0;
  g0[0] = 1u;                                        // count=1, user mode
  g0[1] = loff;                                      // lds_addr [63:32]
  g0[2] = (unsigned)(ga & 0xffffffffu);              // global_addr [95:64]
  g0[3] = (unsigned)((ga >> 32) & 0x01ffffffu)       // global_addr [120:96]
          | 0x80000000u;                             // type=2 ("image") [127:126]
  v8i g1;
  g1[0] = (int)(1u << 16);                           // data_size=1 (2B), mask=0
  g1[1] = (int)((tensor_d0 & 0xffffu) << 16);        // tensor_dim0 [79:48] lo
  g1[2] = (int)(((tensor_d0 >> 16) & 0xffffu) |      // tensor_dim0 hi
                ((tensor_d1 & 0xffffu) << 16));      // tensor_dim1 [111:80] lo
  g1[3] = (int)(((tensor_d1 >> 16) & 0xffffu) |      // tensor_dim1 hi
                ((tile_d0 & 0xffffu) << 16));        // tile_dim0 [127:112]
  g1[4] = (int)(tile_d1 & 0xffffu);                  // tile_dim1 [143:128], dim2=0
  g1[5] = (int)stride;                               // tensor_dim0_stride [191:160]
  g1[6] = 0;                                         // stride hi / dim1_stride lo
  g1[7] = 0;
  v4i z4 = {0, 0, 0, 0};
  v8i z8 = {0, 0, 0, 0, 0, 0, 0, 0};
  __builtin_amdgcn_tensor_load_to_lds(g0, g1, z4, z4, z8, 0);
}
#endif

// ---------------------------------------------------------------- fp32 -> bf16
__global__ __launch_bounds__(256) void cvt_bf16_kernel(const float* __restrict__ src,
                                                       u16* __restrict__ dst, int n) {
  int i = blockIdx.x * blockDim.x + threadIdx.x;
  int stride = gridDim.x * blockDim.x;
  for (; i < n; i += stride) dst[i] = f2bf(src[i]);
}

// ------------------------------------------------- QKV projection (bf16 WMMA)
// grid = (BN/128, D/64, 3), block = 256 (8 waves). z: 0=Q 1=K 2=V
// 128x64 block tile, K-step 64, double-buffered LDS fed by the TDM.
__global__ __launch_bounds__(256) void qkv_gemm_kernel(
    const u16* __restrict__ Xbf, const u16* __restrict__ Wbf,
    const float* __restrict__ bq, const float* __restrict__ bk,
    const float* __restrict__ bv,
    u16* __restrict__ Qh, u16* __restrict__ Kh, u16* __restrict__ Vt) {
  const int which = blockIdx.z;
  const u16*   W    = Wbf + (size_t)which * D_ * D_;
  const float* bias = (which == 0) ? bq : (which == 1) ? bk : bv;

  const int m0 = blockIdx.x * 128;
  const int n0 = blockIdx.y * 64;
  const int tid  = threadIdx.x;
  const int wave = tid >> 5;
  const int lane = tid & 31;
  const int mrow  = lane & 15;
  const int half8 = (lane < 16) ? 0 : 8;   // A-frag K sub-chunk
  const int kb16  = (lane < 16) ? 0 : 16;  // B-frag K half

  __shared__ __align__(16) u16 sA[2][128 * 64];   // 2 x 16 KB
  __shared__ __align__(16) u16 sB[2][64 * 64];    // 2 x  8 KB

  v8f acc[4] = {};

  auto compute = [&](int buf) {
    const u16* Arow = sA[buf] + (wave * 16 + mrow) * 64;
    const v16bf af0 = ldfragA(Arow,      half8);
    const v16bf af1 = ldfragA(Arow + 32, half8);
#pragma unroll
    for (int nt = 0; nt < 4; ++nt) {
      const u16* Brow = sB[buf] + (nt * 16 + mrow) * 64;
      const v16bf b0 = ldfrag16(Brow + kb16);
      acc[nt] = __builtin_amdgcn_wmma_f32_16x16x32_bf16(
          false, af0, false, b0, (short)0, acc[nt], false, false);
      const v16bf b1 = ldfrag16(Brow + 32 + kb16);
      acc[nt] = __builtin_amdgcn_wmma_f32_16x16x32_bf16(
          false, af1, false, b1, (short)0, acc[nt], false, false);
    }
  };

#if USE_TDM
  auto issue_stage = [&](int buf, int k0) {
    if (wave == 0) {
      tdm_load_2d(lds_off(sA[buf]), Xbf + (size_t)m0 * D_ + k0,
                  64, 128, D_, BN_, D_);
      tdm_load_2d(lds_off(sB[buf]), W + (size_t)n0 * D_ + k0,
                  64, 64, D_, D_, D_);
    }
  };
  issue_stage(0, 0);
  issue_stage(1, 64);
  int buf = 0;
  for (int k0 = 0; k0 < D_; k0 += 64, buf ^= 1) {
    if (wave == 0) {
      if (k0 + 64 < D_) __builtin_amdgcn_s_wait_tensorcnt((short)2);
      else              __builtin_amdgcn_s_wait_tensorcnt((short)0);
    }
    __syncthreads();          // publish TDM-filled LDS to all waves
    compute(buf);
    __syncthreads();          // all waves done reading this buffer
    if (k0 + 128 < D_) issue_stage(buf, k0 + 128);
  }
#else
  for (int k0 = 0; k0 < D_; k0 += 64) {
    {
      const int r = tid >> 1, c = (tid & 1) << 5;          // A: 32 elems/thread
      const u16* ga = Xbf + (size_t)(m0 + r) * D_ + k0 + c;
      uint4* da = (uint4*)(sA[0] + r * 64 + c);
#pragma unroll
      for (int q = 0; q < 4; ++q) da[q] = ((const uint4*)ga)[q];
      const int rb = tid >> 2, cb = (tid & 3) << 4;        // B: 16 elems/thread
      const u16* gb = W + (size_t)(n0 + rb) * D_ + k0 + cb;
      uint4* db = (uint4*)(sB[0] + rb * 64 + cb);
      db[0] = ((const uint4*)gb)[0];
      db[1] = ((const uint4*)gb)[1];
    }
    __syncthreads();
    compute(0);
    __syncthreads();
  }
#endif

  // epilogue: bias add, Q pre-scale, head-split store (V stored transposed)
  const int col16 = lane & 15;
#pragma unroll
  for (int nt = 0; nt < 4; ++nt) {
    const int coln = n0 + nt * 16 + col16;
    const int h  = coln >> 6;
    const int dk = coln & 63;
    const float bb = bias[coln];
#pragma unroll
    for (int v = 0; v < 8; ++v) {
      const int rg = m0 + wave * 16 + v + half8;   // row in [0, BN)
      const int b  = rg >> 11;                     // / N_
      const int n  = rg & (N_ - 1);
      float val = acc[nt][v] + bb;
      if (which == 0) val *= RSCALE;
      const u16 o = f2bf(val);
      if (which == 2)
        Vt[((size_t)(b * H_ + h) * DK_ + dk) * N_ + n] = o;         // [b,h,dk,n]
      else if (which == 0)
        Qh[((size_t)(b * H_ + h) * N_ + n) * DK_ + dk] = o;         // [b,h,n,dk]
      else
        Kh[((size_t)(b * H_ + h) * N_ + n) * DK_ + dk] = o;
    }
  }
}

// -------------------------------------------- flash attention (bf16 WMMA core)
// grid = (N/64, H, B), block = 128 (4 independent waves of 16 query rows),
// 64-key blocks: 16 WMMAs per iteration.
__global__ __launch_bounds__(128) void flash_attn_kernel(
    const u16* __restrict__ Qh, const u16* __restrict__ Kh,
    const u16* __restrict__ Vt, const int* __restrict__ xmask,
    const int* __restrict__ dirp, float* __restrict__ out) {
  const int b = blockIdx.z, h = blockIdx.y;
  const int wave = threadIdx.x >> 5;
  const int lane = threadIdx.x & 31;
  const int q0 = blockIdx.x * 64 + wave * 16;
  const int dir = *dirp;

  const u16* Qb = Qh + (size_t)(b * H_ + h) * N_ * DK_;
  const u16* Kb = Kh + (size_t)(b * H_ + h) * N_ * DK_;
  const u16* Vb = Vt + (size_t)(b * H_ + h) * DK_ * N_;

  const int mrow  = lane & 15;
  const int half8 = (lane < 16) ? 0 : 8;
  const int kb16  = (lane < 16) ? 0 : 16;
  const int col16 = lane & 15;

  // Q fragments for this wave's 16 rows (dk chunks 0..31, 32..63), pre-scaled
  v16bf qa[2];
  {
    const u16* qrow = Qb + (size_t)(q0 + mrow) * DK_;
#pragma unroll
    for (int c = 0; c < 2; ++c) qa[c] = ldfragA(qrow + c * 32, half8);
  }

  v8f zacc[4] = {};
  float rmax[8], rsum[8];
#pragma unroll
  for (int v = 0; v < 8; ++v) { rmax[v] = -3.0e38f; rsum[v] = 0.0f; }

  __shared__ __align__(16) u16 lP[4][16 * 64];   // wave-private P tiles (8 KB)
  u16* myP = lP[wave];

  for (int j0 = 0; j0 < N_; j0 += 64) {
    if (j0 + 64 < N_) {
      __builtin_prefetch(Kb + (size_t)(j0 + 64 + mrow) * DK_, 0, 1);
      __builtin_prefetch(Vb + (size_t)mrow * N_ + j0 + 64, 0, 1);
    }
    // S = Q @ K^T : four 16x16 score tiles, K-dim 64 in two WMMA steps each
    v8f sacc[4] = {};
#pragma unroll
    for (int j = 0; j < 4; ++j) {
      const u16* krow = Kb + (size_t)(j0 + j * 16 + mrow) * DK_;
#pragma unroll
      for (int c = 0; c < 2; ++c) {
        const v16bf kf = ldfrag16(krow + c * 32 + kb16);
        sacc[j] = __builtin_amdgcn_wmma_f32_16x16x32_bf16(
            false, qa[c], false, kf, (short)0, sacc[j], false, false);
      }
    }

    // mask + online softmax (each row lives across one 16-lane half)
#pragma unroll
    for (int v = 0; v < 8; ++v) {
      const int qg = q0 + v + half8;
      float s[4], p[4];
#pragma unroll
      for (int j = 0; j < 4; ++j) {
        const int ki = j0 + j * 16 + col16;
        const size_t mi = dir ? ((size_t)(b * N_ + qg) * N_ + ki)
                              : ((size_t)(b * N_ + ki) * N_ + qg);
        s[j] = (xmask[mi] == 0) ? NEGBIG : sacc[j][v];
      }
      float bmax = fmaxf(fmaxf(s[0], s[1]), fmaxf(s[2], s[3]));
#pragma unroll
      for (int off = 1; off < 16; off <<= 1)
        bmax = fmaxf(bmax, __shfl_xor(bmax, off, 32));
      const float mnew = fmaxf(rmax[v], bmax);
      const float f = __expf(rmax[v] - mnew);
      float ps = 0.0f;
#pragma unroll
      for (int j = 0; j < 4; ++j) { p[j] = __expf(s[j] - mnew); ps += p[j]; }
#pragma unroll
      for (int off = 1; off < 16; off <<= 1)
        ps += __shfl_xor(ps, off, 32);
      rsum[v] = rsum[v] * f + ps;
      rmax[v] = mnew;
#pragma unroll
      for (int nt = 0; nt < 4; ++nt) zacc[nt][v] *= f;
#pragma unroll
      for (int j = 0; j < 4; ++j)
        myP[(v + half8) * 64 + j * 16 + col16] = f2bf(p[j]);
    }

    // Z += P @ V  (P re-fetched from LDS in A-fragment layout; V transposed)
#pragma unroll
    for (int kc = 0; kc < 2; ++kc) {
      const v16bf pa = ldfragA(myP + mrow * 64 + kc * 32, half8);
#pragma unroll
      for (int nt = 0; nt < 4; ++nt) {
        const u16* vrow = Vb + (size_t)(nt * 16 + mrow) * N_ + j0 + kc * 32;
        const v16bf vf = ldfrag16(vrow + kb16);
        zacc[nt] = __builtin_amdgcn_wmma_f32_16x16x32_bf16(
            false, pa, false, vf, (short)0, zacc[nt], false, false);
      }
    }
  }

  // normalize + store z in [B, N, D] layout
#pragma unroll
  for (int nt = 0; nt < 4; ++nt) {
    const int dk = nt * 16 + col16;
#pragma unroll
    for (int v = 0; v < 8; ++v) {
      const int qr = q0 + v + half8;
      out[((size_t)b * N_ + qr) * D_ + h * DK_ + dk] = zacc[nt][v] / rsum[v];
    }
  }
}

// ------------------------------------------------------------------- launcher
extern "C" void kernel_launch(void* const* d_in, const int* in_sizes, int n_in,
                              void* d_out, int out_size, void* d_ws, size_t ws_size,
                              hipStream_t stream) {
  const float* x    = (const float*)d_in[0];
  const int*   mask = (const int*)d_in[1];
  const int*   dirp = (const int*)d_in[2];
  const float* Wq   = (const float*)d_in[3];
  const float* bq   = (const float*)d_in[4];
  const float* Wk   = (const float*)d_in[5];
  const float* bk   = (const float*)d_in[6];
  const float* Wv   = (const float*)d_in[7];
  const float* bv   = (const float*)d_in[8];

  u16* Xbf = (u16*)d_ws;                       // BN * D
  u16* Wbf = Xbf + (size_t)BN_ * D_;           // 3 * D * D
  u16* Qh  = Wbf + (size_t)3 * D_ * D_;        // B*H*N*DK
  u16* Kh  = Qh  + (size_t)B_ * H_ * N_ * DK_;
  u16* Vt  = Kh  + (size_t)B_ * H_ * N_ * DK_;

  cvt_bf16_kernel<<<4096, 256, 0, stream>>>(x, Xbf, BN_ * D_);
  cvt_bf16_kernel<<<1024, 256, 0, stream>>>(Wq, Wbf,            D_ * D_);
  cvt_bf16_kernel<<<1024, 256, 0, stream>>>(Wk, Wbf + D_ * D_,  D_ * D_);
  cvt_bf16_kernel<<<1024, 256, 0, stream>>>(Wv, Wbf + 2*D_*D_,  D_ * D_);

  dim3 g1(BN_ / 128, D_ / 64, 3);
  qkv_gemm_kernel<<<g1, 256, 0, stream>>>(Xbf, Wbf, bq, bk, bv, Qh, Kh, Vt);

  dim3 g2(N_ / 64, H_, B_);
  flash_attn_kernel<<<g2, 128, 0, stream>>>(Qh, Kh, Vt, mask, dirp, (float*)d_out);
}
